// OLN_RPN_67010079752796
// MI455X (gfx1250) — matrix operand order repaired and checked by reading, not compiled, not run on hardware
//
#include <hip/hip_runtime.h>
#include <hip/hip_bf16.h>
#include <math.h>

typedef __attribute__((ext_vector_type(16))) __bf16 v16bf;
typedef __attribute__((ext_vector_type(8)))  float  v8f;
typedef __attribute__((ext_vector_type(4)))  unsigned int v4u;
typedef __attribute__((ext_vector_type(8)))  int v8i;
typedef __attribute__((ext_vector_type(4)))  int v4i;

#define NLVL 5
#define BATCH 2
#define CCH 256
#define PRE_K 1000
#define POST_K 1000
#define CAND_TOT 4768           // 4*1000 + 768
#define POS_TOT 87296           // sum H*W over levels
#define NTOT 261888             // 3 * POS_TOT
#define NMS_TH 0.7f
#define IMGSZ 1024.0f
#define SCALE_CLAMP 4.135166556742356f   // log(1000/16)

__device__ const int d_kL[NLVL]     = {1000,1000,1000,1000,768};
__device__ const int d_cOff[NLVL]   = {0,1000,2000,3000,4000};
__device__ const int d_posOff[NLVL] = {0,65536,81920,86016,87040};

__device__ __forceinline__ unsigned int f2bf(float f) {
    unsigned int u = __float_as_uint(f);
    return ((u + 0x7FFFu + ((u >> 16) & 1u)) >> 16) & 0xFFFFu;  // RNE
}
__device__ __forceinline__ float bf2f(unsigned int h) {
    return __uint_as_float(h << 16);
}
__device__ __forceinline__ unsigned int ordkey(float s) {
    unsigned int u = __float_as_uint(s);
    return (u & 0x80000000u) ? ~u : (u | 0x80000000u);
}

// ---- Tensor Data Mover: load a 64-row x 256-col bf16 tile (row stride 256)
// into LDS with 1 dword of padding every 128 dwords (pitch 129 dwords).
__device__ __forceinline__ void tdm_load_A(const unsigned short* gsrc, unsigned ldsOff) {
    unsigned long long ga = (unsigned long long)(uintptr_t)gsrc;
    v4u g0;
    g0[0] = 1u;                                             // count=1, user mode
    g0[1] = ldsOff;                                         // lds_addr (bytes)
    g0[2] = (unsigned)(ga & 0xFFFFFFFFull);                 // global_addr[31:0]
    g0[3] = (unsigned)((ga >> 32) & 0x1FFFFFFull) | (2u << 30); // addr[56:32] | type=2
    v8i g1;
    g1[0] = (int)((1u << 16) | (1u << 20) | (6u << 22));    // data_size=2B, pad_en, pad_interval=128DW, pad_amount=1DW
    g1[1] = (int)(256u << 16);                              // tensor_dim0 = 256 (lo16)
    g1[2] = (int)(64u << 16);                               // dim0 hi=0 | tensor_dim1 lo16 = 64
    g1[3] = (int)(256u << 16);                              // dim1 hi=0 | tile_dim0 = 256
    g1[4] = 64;                                             // tile_dim1 = 64, tile_dim2 = 0
    g1[5] = 256;                                            // tensor_dim0_stride lo32 = 256
    g1[6] = 0;                                              // stride hi | dim1_stride lo
    g1[7] = 0;
    v4i z4 = {0, 0, 0, 0};
    v8i z8 = {0, 0, 0, 0, 0, 0, 0, 0};
    __builtin_amdgcn_tensor_load_to_lds(g0, g1, z4, z4, z8, 0);
}

// ---------------------------------------------------------------- weights prep
__global__ void prep_weights(const float* __restrict__ cw, unsigned short* __restrict__ wbf) {
    int g = blockIdx.x * blockDim.x + threadIdx.x;
    if (g >= 9 * CCH * CCH) return;
    int ci = g & 255;
    int co = (g >> 8) & 255;
    int t9 = g >> 16;                       // 256*256 == 65536
    float v = cw[(co * CCH + ci) * 9 + t9]; // [co][ci][ky][kx]
    wbf[(t9 * CCH + co) * CCH + ci] = (unsigned short)f2bf(v);
}

// ------------------------------------------------------- 3x3 conv + ReLU (WMMA)
// block: 64 out-channels x 64 spatial, 256 threads = 8 waves.
// wave (mw,nw): mw in 0..3 -> 16 channels, nw in 0..1 -> 32 spatial (2 WMMA N-tiles).
__global__ __launch_bounds__(256)
void conv3x3_wmma(const float* __restrict__ f, const float* __restrict__ convb,
                  const unsigned short* __restrict__ wbf, unsigned short* __restrict__ tArr,
                  int H, int W, int posOff)
{
    __shared__ unsigned int Asm[64 * 129];   // weights tile   [m][kword], pitch 129
    __shared__ unsigned int Bsm[64 * 129];   // im2col tile    [n][kword], pitch 129

    const int tid = threadIdx.x;
    const int b  = blockIdx.z;
    const int coBase = blockIdx.y * 64;
    const int sBase  = blockIdx.x * 64;
    const int HW = H * W;
    const int wave = tid >> 5, lane = tid & 31;
    const int mw = wave & 3, nw = wave >> 2;

    // B loader mapping: spatial = (tid&31) + 32*sh, channels bkq*32..+31
    const int bs = tid & 31, bkq = tid >> 5;
    int py[2], px[2]; bool pv[2];
    #pragma unroll
    for (int sh = 0; sh < 2; ++sh) {
        const int pos = sBase + bs + 32 * sh;
        py[sh] = pos / W; px[sh] = pos % W; pv[sh] = pos < HW;
    }

    const unsigned ldsA = (unsigned)(uintptr_t)(void*)Asm;
    const size_t fbase = (size_t)b * CCH * HW;

    v8f acc0 = {}, acc1 = {};

    for (int t9 = 0; t9 < 9; ++t9) {
        const int ky = t9 / 3 - 1, kx = t9 % 3 - 1;
        __syncthreads();                    // prev-stage LDS reads complete
        if (wave == 0) {                    // wave-uniform: one TDM issue per block
            tdm_load_A(wbf + ((size_t)t9 * 256 + coBase) * 256, ldsA);
        }
        // im2col gather: f32 NCHW -> bf16 LDS [spatial][channel]
        #pragma unroll
        for (int sh = 0; sh < 2; ++sh) {
            const int yy = py[sh] + ky, xx = px[sh] + kx;
            const bool bval = pv[sh] && ((unsigned)yy < (unsigned)H) && ((unsigned)xx < (unsigned)W);
            const float* fb = f + fbase + (size_t)yy * W + xx + (size_t)(bkq * 32) * HW;
            const int srow = (bs + 32 * sh) * 129 + bkq * 16;
            #pragma unroll
            for (int j = 0; j < 16; ++j) {
                float v0 = 0.f, v1 = 0.f;
                if (bval) { v0 = fb[(size_t)(2 * j) * HW]; v1 = fb[(size_t)(2 * j + 1) * HW]; }
                Bsm[srow + j] = f2bf(v0) | (f2bf(v1) << 16);
            }
        }
        if (wave == 0) __builtin_amdgcn_s_wait_tensorcnt(0);
        __syncthreads();

        const int hm = lane >> 4;
        const int ar  = mw * 16 + (lane & 15);
        const int bc0 = nw * 32 + (lane & 15);
        #pragma unroll
        for (int kc = 0; kc < 8; ++kc) {
            union { v16bf v; unsigned int u[8]; } Af, Bf0, Bf1;
            #pragma unroll
            for (int v = 0; v < 8; ++v) {
                const int aw = kc * 16 + v + hm * 8;
                Af.u[v]  = Asm[ar * 129 + aw];
                const int bw = kc * 16 + v + hm * 8;
                Bf0.u[v] = Bsm[bc0 * 129 + bw];
                Bf1.u[v] = Bsm[(bc0 + 16) * 129 + bw];
            }
            acc0 = __builtin_amdgcn_wmma_f32_16x16x32_bf16(
                       false, Af.v, false, Bf0.v, (short)0, acc0, false, false);
            acc1 = __builtin_amdgcn_wmma_f32_16x16x32_bf16(
                       false, Af.v, false, Bf1.v, (short)0, acc1, false, false);
        }
    }

    // epilogue: bias + ReLU -> bf16 NHWC, one 16B store per lane per N-tile
    const int co = coBase + mw * 16 + (lane >> 4) * 8;
    float bias[8];
    #pragma unroll
    for (int r = 0; r < 8; ++r) bias[r] = convb[co + r];
    #pragma unroll
    for (int half = 0; half < 2; ++half) {
        const int n = sBase + nw * 32 + (lane & 15) + 16 * half;
        if (n < HW) {
            const v8f a = half ? acc1 : acc0;
            unsigned int outw[4];
            #pragma unroll
            for (int r = 0; r < 4; ++r) {
                float a0 = a[2 * r]     + bias[2 * r];
                float a1 = a[2 * r + 1] + bias[2 * r + 1];
                a0 = a0 > 0.f ? a0 : 0.f;
                a1 = a1 > 0.f ? a1 : 0.f;
                outw[r] = f2bf(a0) | (f2bf(a1) << 16);
            }
            const size_t tbase = ((size_t)b * POS_TOT + posOff + n) * 256 + co;
            *((uint4*)(tArr + tbase)) = make_uint4(outw[0], outw[1], outw[2], outw[3]);
        }
    }
}

// ----------------------------------------------------------------- 1x1 heads
__global__ void head_1x1(const unsigned short* __restrict__ tArr,
                         const float* __restrict__ objw, const float* __restrict__ objb,
                         const float* __restrict__ delw, const float* __restrict__ delb,
                         float* __restrict__ scores, float* __restrict__ deltas,
                         int HW, int posOff, int scOff)
{
    int g = blockIdx.x * blockDim.x + threadIdx.x;
    if (g >= BATCH * HW * 16) return;
    const int oi = g & 15;
    if (oi >= 15) return;
    const int pe = g >> 4;
    const int b = pe / HW, pos = pe % HW;
    const float* wv = (oi < 3) ? (objw + oi * CCH) : (delw + (oi - 3) * CCH);
    const uint4* tp = (const uint4*)(tArr + ((size_t)b * POS_TOT + posOff + pos) * 256);
    float acc = 0.f;
    #pragma unroll 4
    for (int i = 0; i < 32; ++i) {
        uint4 q = tp[i];
        const float4* wp = (const float4*)(wv + i * 8);
        float4 w0 = wp[0], w1 = wp[1];
        acc += bf2f(q.x & 0xffffu) * w0.x + bf2f(q.x >> 16) * w0.y
             + bf2f(q.y & 0xffffu) * w0.z + bf2f(q.y >> 16) * w0.w
             + bf2f(q.z & 0xffffu) * w1.x + bf2f(q.z >> 16) * w1.y
             + bf2f(q.w & 0xffffu) * w1.z + bf2f(q.w >> 16) * w1.w;
    }
    if (oi < 3) {
        scores[(size_t)b * NTOT + scOff + pos * 3 + oi] = acc + objb[oi];
    } else {
        const int o = oi - 3, a = o >> 2, cc = o & 3;
        deltas[(((size_t)b * NTOT + scOff + pos * 3 + a) << 2) + cc] = acc + delb[o];
    }
}

// ----------------------------------------------------------------- top-k path
__global__ void zero_u32(unsigned int* __restrict__ p, int n) {
    int g = blockIdx.x * blockDim.x + threadIdx.x;
    if (g < n) p[g] = 0u;
}

__global__ void hist_kernel(const float* __restrict__ scores, unsigned int* __restrict__ hist,
                            int n, int scOff, int lvl)
{
    const int b = blockIdx.y;
    unsigned int* h = hist + (lvl * 2 + b) * 2048;
    for (int i = blockIdx.x * blockDim.x + threadIdx.x; i < n; i += gridDim.x * blockDim.x) {
        float s = scores[(size_t)b * NTOT + scOff + i];
        atomicAdd(&h[ordkey(s) >> 21], 1u);
    }
}

__global__ void scan_kernel(const unsigned int* __restrict__ hist, int* __restrict__ thr) {
    const int lb = blockIdx.x;          // 0..9 = lvl*2+b
    if (threadIdx.x != 0) return;
    const int k = d_kL[lb >> 1];
    const unsigned int* h = hist + lb * 2048;
    int acc = 0, T = 0, above = 0;
    for (int bin = 2047; bin >= 0; --bin) {
        const int c = (int)h[bin];
        if (acc + c >= k) { T = bin; above = acc; break; }
        acc += c;
    }
    thr[lb * 3 + 0] = T;
    thr[lb * 3 + 1] = above;
    thr[lb * 3 + 2] = k - above;
}

__global__ void compact_kernel(const float* __restrict__ scores, const int* __restrict__ thr,
                               unsigned int* __restrict__ ctrs,
                               int* __restrict__ selIdx, float* __restrict__ selScore,
                               int n, int scOff, int lvl)
{
    const int b = blockIdx.y;
    const int lb = lvl * 2 + b;
    const int T = thr[lb * 3], above = thr[lb * 3 + 1], tieNeed = thr[lb * 3 + 2];
    for (int i = blockIdx.x * blockDim.x + threadIdx.x; i < n; i += gridDim.x * blockDim.x) {
        const float s = scores[(size_t)b * NTOT + scOff + i];
        const int bin = (int)(ordkey(s) >> 21);
        if (bin > T) {
            unsigned int p = atomicAdd(&ctrs[lb * 2], 1u);
            selIdx[lb * PRE_K + p] = i;
            selScore[lb * PRE_K + p] = s;
        } else if (bin == T) {
            unsigned int tp = atomicAdd(&ctrs[lb * 2 + 1], 1u);
            if ((int)tp < tieNeed) {
                const int q = above + (int)tp;
                selIdx[lb * PRE_K + q] = i;
                selScore[lb * PRE_K + q] = s;
            }
        }
    }
}

// ------------------------------------------------------ decode boxes per cand
__global__ void build_cand(const int* __restrict__ selIdx, const float* __restrict__ selScore,
                           const float* __restrict__ deltas,
                           const float* a0, const float* a1, const float* a2,
                           const float* a3, const float* a4,
                           float* __restrict__ cbox, float* __restrict__ csc,
                           float* __restrict__ clv)
{
    int g = blockIdx.x * blockDim.x + threadIdx.x;
    if (g >= BATCH * NLVL * PRE_K) return;
    const int b = g / (NLVL * PRE_K);
    const int r = g % (NLVL * PRE_K);
    const int lvl = r / PRE_K, j = r % PRE_K;
    if (j >= d_kL[lvl]) return;
    const int lb = lvl * 2 + b;
    const int idx = selIdx[lb * PRE_K + j];
    float sc = selScore[lb * PRE_K + j];
    const int scOff = d_posOff[lvl] * 3;
    const float* dp = deltas + (((size_t)b * NTOT + scOff + idx) << 2);
    const float* ap = (lvl == 0) ? a0 : (lvl == 1) ? a1 : (lvl == 2) ? a2 : (lvl == 3) ? a3 : a4;
    ap += (size_t)idx * 4;
    const float ax1 = ap[0], ay1 = ap[1], ax2 = ap[2], ay2 = ap[3];
    const float wa = ax2 - ax1, ha = ay2 - ay1;
    const float cxa = ax1 + 0.5f * wa, cya = ay1 + 0.5f * ha;
    const float dx = dp[0], dy = dp[1];
    const float dw = fminf(dp[2], SCALE_CLAMP), dh = fminf(dp[3], SCALE_CLAMP);
    const float cx = dx * wa + cxa, cy = dy * ha + cya;
    const float w = expf(dw) * wa, h = expf(dh) * ha;
    const float x1 = fminf(fmaxf(cx - 0.5f * w, 0.f), IMGSZ);
    const float y1 = fminf(fmaxf(cy - 0.5f * h, 0.f), IMGSZ);
    const float x2 = fminf(fmaxf(cx + 0.5f * w, 0.f), IMGSZ);
    const float y2 = fminf(fmaxf(cy + 0.5f * h, 0.f), IMGSZ);
    if (!((x2 > x1) && (y2 > y1))) sc = -1e9f;
    const int ci = b * CAND_TOT + d_cOff[lvl] + j;
    cbox[ci * 4 + 0] = x1; cbox[ci * 4 + 1] = y1;
    cbox[ci * 4 + 2] = x2; cbox[ci * 4 + 3] = y2;
    csc[ci] = sc;
    clv[ci] = (float)lvl;
}

// --------------------------------------------------------------------- NMS
__global__ __launch_bounds__(1024)
void nms_kernel(const float* __restrict__ cbox, const float* __restrict__ csc,
                const float* __restrict__ clv, float* __restrict__ out)
{
    extern __shared__ float sm[];
    float* X1 = sm;
    float* Y1 = X1 + CAND_TOT;
    float* X2 = Y1 + CAND_TOT;
    float* Y2 = X2 + CAND_TOT;
    float* AR = Y2 + CAND_TOT;
    float* SC = AR + CAND_TOT;
    float* redV = SC + CAND_TOT;         // 1024
    int*   redI = (int*)(redV + 1024);   // 1024

    const int tid = threadIdx.x, b = blockIdx.x;
    for (int i = tid; i < CAND_TOT; i += 1024) {
        const float off = clv[b * CAND_TOT + i] * (2.0f * IMGSZ);
        const float x1 = cbox[(b * CAND_TOT + i) * 4 + 0] + off;
        const float y1 = cbox[(b * CAND_TOT + i) * 4 + 1] + off;
        const float x2 = cbox[(b * CAND_TOT + i) * 4 + 2] + off;
        const float y2 = cbox[(b * CAND_TOT + i) * 4 + 3] + off;
        X1[i] = x1; Y1[i] = y1; X2[i] = x2; Y2[i] = y2;
        AR[i] = (x2 - x1) * (y2 - y1);
        SC[i] = csc[b * CAND_TOT + i];
    }
    __syncthreads();

    for (int it = 0; it < POST_K; ++it) {
        float mv = -3e38f; int mi = 0;
        for (int i = tid; i < CAND_TOT; i += 1024) {
            const float s = SC[i];
            if (s > mv || (s == mv && i < mi)) { mv = s; mi = i; }
        }
        redV[tid] = mv; redI[tid] = mi;
        __syncthreads();
        for (int s = 512; s > 0; s >>= 1) {
            if (tid < s) {
                const float v2 = redV[tid + s]; const int i2 = redI[tid + s];
                if (v2 > redV[tid] || (v2 == redV[tid] && i2 < redI[tid])) {
                    redV[tid] = v2; redI[tid] = i2;
                }
            }
            __syncthreads();
        }
        const int j = redI[0];
        const float sj = redV[0];
        if (tid == 0) {
            const bool ok = sj > -1e8f;
            float* o = out + ((size_t)b * POST_K + it) * 5;
            const float* bj = cbox + ((size_t)b * CAND_TOT + j) * 4;
            o[0] = ok ? bj[0] : 0.f;
            o[1] = ok ? bj[1] : 0.f;
            o[2] = ok ? bj[2] : 0.f;
            o[3] = ok ? bj[3] : 0.f;
            o[4] = ok ? sj : -1e9f;
        }
        const float bx1 = X1[j], by1 = Y1[j], bx2 = X2[j], by2 = Y2[j], ba = AR[j];
        __syncthreads();
        for (int i = tid; i < CAND_TOT; i += 1024) {
            const float xx1 = fmaxf(bx1, X1[i]), yy1 = fmaxf(by1, Y1[i]);
            const float xx2 = fminf(bx2, X2[i]), yy2 = fminf(by2, Y2[i]);
            const float inter = fmaxf(xx2 - xx1, 0.f) * fmaxf(yy2 - yy1, 0.f);
            const float iou = inter / (ba + AR[i] - inter + 1e-9f);
            if (iou > NMS_TH) SC[i] = -1e9f;
        }
        __syncthreads();
    }
}

// ------------------------------------------------------------------- launch
extern "C" void kernel_launch(void* const* d_in, const int* in_sizes, int n_in,
                              void* d_out, int out_size, void* d_ws, size_t ws_size,
                              hipStream_t stream)
{
    (void)in_sizes; (void)n_in; (void)out_size; (void)ws_size;
    const float* fL[NLVL] = {(const float*)d_in[0], (const float*)d_in[2], (const float*)d_in[4],
                             (const float*)d_in[6], (const float*)d_in[8]};
    const float* aL[NLVL] = {(const float*)d_in[1], (const float*)d_in[3], (const float*)d_in[5],
                             (const float*)d_in[7], (const float*)d_in[9]};
    const float* conv_w  = (const float*)d_in[10];
    const float* conv_b  = (const float*)d_in[11];
    const float* obj_w   = (const float*)d_in[12];
    const float* obj_b   = (const float*)d_in[13];
    const float* delta_w = (const float*)d_in[14];
    const float* delta_b = (const float*)d_in[15];
    float* out = (float*)d_out;

    static const int h_H[NLVL]      = {256, 128, 64, 32, 16};
    static const int h_HW[NLVL]     = {65536, 16384, 4096, 1024, 256};
    static const int h_posOff[NLVL] = {0, 65536, 81920, 86016, 87040};

    // workspace carve-up (256B aligned)
    char* ws = (char*)d_ws;
    size_t off = 0;
    auto carve = [&](size_t bytes) { size_t o = off; off = (off + bytes + 255) & ~(size_t)255; return o; };
    unsigned short* wbf = (unsigned short*)(ws + carve((size_t)9 * CCH * CCH * 2));
    unsigned short* tArr = (unsigned short*)(ws + carve((size_t)BATCH * POS_TOT * 256 * 2));
    float* scores = (float*)(ws + carve((size_t)BATCH * NTOT * 4));
    float* deltas = (float*)(ws + carve((size_t)BATCH * NTOT * 4 * 4));
    unsigned int* hist = (unsigned int*)(ws + carve((size_t)10 * 2048 * 4));
    unsigned int* ctrs = (unsigned int*)(ws + carve((size_t)10 * 2 * 4));
    int* thr = (int*)(ws + carve((size_t)10 * 3 * 4));
    int* selIdx = (int*)(ws + carve((size_t)10 * PRE_K * 4));
    float* selScore = (float*)(ws + carve((size_t)10 * PRE_K * 4));
    float* cbox = (float*)(ws + carve((size_t)BATCH * CAND_TOT * 4 * 4));
    float* csc = (float*)(ws + carve((size_t)BATCH * CAND_TOT * 4));
    float* clv = (float*)(ws + carve((size_t)BATCH * CAND_TOT * 4));

    // 1) clear histogram + counters
    zero_u32<<<(10 * 2048 + 255) / 256, 256, 0, stream>>>(hist, 10 * 2048);
    zero_u32<<<1, 32, 0, stream>>>(ctrs, 20);

    // 2) weights -> bf16 (TDM-friendly [tap][co][ci] layout)
    prep_weights<<<(9 * CCH * CCH + 255) / 256, 256, 0, stream>>>(conv_w, wbf);

    // 3) conv3x3 + ReLU via WMMA (TDM weight staging), then 1x1 heads
    for (int l = 0; l < NLVL; ++l) {
        const int HW = h_HW[l];
        dim3 g((HW + 63) / 64, 4, BATCH);
        conv3x3_wmma<<<g, 256, 0, stream>>>(fL[l], conv_b, wbf, tArr, h_H[l], h_H[l], h_posOff[l]);
    }
    for (int l = 0; l < NLVL; ++l) {
        const int HW = h_HW[l];
        head_1x1<<<(BATCH * HW * 16 + 255) / 256, 256, 0, stream>>>(
            tArr, obj_w, obj_b, delta_w, delta_b, scores, deltas,
            HW, h_posOff[l], h_posOff[l] * 3);
    }

    // 4) top-k: histogram -> threshold scan -> compaction
    for (int l = 0; l < NLVL; ++l) {
        const int n = h_HW[l] * 3;
        int gx = (n + 255) / 256; if (gx > 512) gx = 512;
        hist_kernel<<<dim3(gx, BATCH), 256, 0, stream>>>(scores, hist, n, h_posOff[l] * 3, l);
    }
    scan_kernel<<<10, 32, 0, stream>>>(hist, thr);
    for (int l = 0; l < NLVL; ++l) {
        const int n = h_HW[l] * 3;
        int gx = (n + 255) / 256; if (gx > 512) gx = 512;
        compact_kernel<<<dim3(gx, BATCH), 256, 0, stream>>>(
            scores, thr, ctrs, selIdx, selScore, n, h_posOff[l] * 3, l);
    }

    // 5) decode candidates
    build_cand<<<(BATCH * NLVL * PRE_K + 255) / 256, 256, 0, stream>>>(
        selIdx, selScore, deltas, aL[0], aL[1], aL[2], aL[3], aL[4], cbox, csc, clv);

    // 6) NMS (one workgroup per image, everything in LDS)
    const size_t smem = (size_t)(6 * CAND_TOT + 1024) * sizeof(float) + 1024 * sizeof(int);
    nms_kernel<<<BATCH, 1024, smem, stream>>>(cbox, csc, clv, out);
}
